// NT_XentLoss_81604378624062
// MI455X (gfx1250) — matrix operand order repaired and checked by reading, not compiled, not run on hardware
//
#include <hip/hip_runtime.h>
#include <hip/hip_bf16.h>
#include <math.h>
#include <stdint.h>

typedef _Float16 v16h __attribute__((ext_vector_type(16)));
typedef _Float16 h8   __attribute__((ext_vector_type(8)));
typedef float    v8f  __attribute__((ext_vector_type(8)));

#define BATCH     4096
#define NROWS     8192          // 2*BATCH
#define DIM       256
#define TEMP_INV  2.0f          // 1/0.5
#define MAXLOGIT  2.0f          // |cos|<=1 -> logits <= 2 (diag = -1e9, excluded)
#define NSPLIT    8
#define NT_TOTAL  (NROWS / 16)          // 512 N-tiles
#define NT_SLICE  (NT_TOTAL / NSPLIT)   // 64 per WG
#define LDSS      264                   // padded LDS row stride in halves

// ---------------- Kernel 1: row-normalize fp32 -> f16 Zn --------------------
__global__ void __launch_bounds__(256) nrm_kernel(const float* __restrict__ zi,
                                                  const float* __restrict__ zj,
                                                  _Float16* __restrict__ zn) {
    const int wid  = threadIdx.x >> 5;
    const int lane = threadIdx.x & 31;
    const int row  = blockIdx.x * 8 + wid;
    const float* src = (row < BATCH) ? (zi + (size_t)row * DIM)
                                     : (zj + (size_t)(row - BATCH) * DIM);
    float4 a = ((const float4*)src)[lane * 2 + 0];
    float4 b = ((const float4*)src)[lane * 2 + 1];
    float ss = a.x*a.x + a.y*a.y + a.z*a.z + a.w*a.w
             + b.x*b.x + b.y*b.y + b.z*b.z + b.w*b.w;
#pragma unroll
    for (int m = 16; m >= 1; m >>= 1) ss += __shfl_xor(ss, m, 32);
    const float scale = 1.0f / fmaxf(sqrtf(ss), 1e-8f);
    h8 o;
    o[0] = (_Float16)(a.x * scale); o[1] = (_Float16)(a.y * scale);
    o[2] = (_Float16)(a.z * scale); o[3] = (_Float16)(a.w * scale);
    o[4] = (_Float16)(b.x * scale); o[5] = (_Float16)(b.y * scale);
    o[6] = (_Float16)(b.z * scale); o[7] = (_Float16)(b.w * scale);
    *(h8*)(zn + (size_t)row * DIM + lane * 8) = o;
}

// ------------- Kernel 2: fused WMMA GEMM + streaming softmax stats ----------
// grid = (64, NSPLIT); 8 waves/WG; wave w owns M-tile rows [ (bx*8+w)*16 , +16 ).
// B tiles double-buffered in LDS, filled with global_load_async_to_lds_b128.
__global__ void __launch_bounds__(256) simk_kernel(const _Float16* __restrict__ zn,
                                                   float* __restrict__ S,
                                                   float* __restrict__ T) {
    __shared__ _Float16 lds[2 * 16 * LDSS];

    const int tid  = threadIdx.x;
    const int wid  = tid >> 5;
    const int lane = tid & 31;
    const int half = lane >> 4;     // 0: lanes 0-15, 1: lanes 16-31
    const int l15  = lane & 15;
    const int m0   = (blockIdx.x * 8 + wid) * 16;

    // ---- Load A tile (16 rows x 256 K) into registers, ISA A-layout:
    // lane<16: K {k0+0..7, k0+16..23}; lane>=16: K {k0+8..15, k0+24..31}
    v16h Areg[8];
    {
        const _Float16* ap = zn + (size_t)(m0 + l15) * DIM + half * 8;
#pragma unroll
        for (int ks = 0; ks < 8; ks++) {
            h8 lo = *(const h8*)(ap + ks * 32);
            h8 hi = *(const h8*)(ap + ks * 32 + 16);
            Areg[ks] = __builtin_shufflevector(lo, hi, 0,1,2,3,4,5,6,7,
                                                        8,9,10,11,12,13,14,15);
        }
    }

    // ---- Per-thread async-staging geometry (2 x b128 per thread = 8KB tile)
    const unsigned ldsbase = (unsigned)(uintptr_t)lds;   // LDS offset (low 32 bits)
    const int r0 = tid >> 5;            // tile row for it=0 (it=1: +8)
    const int c0 = (tid & 31) * 8;      // half offset within row

    auto stage = [&](int buf, int j0) {
#pragma unroll
        for (int it = 0; it < 2; ++it) {
            const int r = r0 + it * 8;
            const unsigned loff = ldsbase +
                (unsigned)(((buf * 16 + r) * LDSS + c0) * 2);
            const unsigned long long ga =
                (unsigned long long)(uintptr_t)(zn + (size_t)(j0 + r) * DIM + c0);
            asm volatile("global_load_async_to_lds_b128 %0, %1, off"
                         :: "v"(loff), "v"(ga) : "memory");
        }
    };

    float sume[8]; float tg[8];
#pragma unroll
    for (int r = 0; r < 8; r++) { sume[r] = 0.0f; tg[r] = 0.0f; }

    const int nt0 = blockIdx.y * NT_SLICE;
    stage(0, nt0 * 16);                       // preload first tile

    for (int nt = 0; nt < NT_SLICE; nt++) {
        const int cur = nt & 1;
        const int j0  = (nt0 + nt) * 16;

        // Tile nt's async loads (issued last iter) landed in LDS:
        asm volatile("s_wait_asynccnt 0x0" ::: "memory");
        __syncthreads();
        // Kick off tile nt+1 into the other buffer; overlaps WMMA below.
        if (nt + 1 < NT_SLICE) stage(cur ^ 1, j0 + 16);

        // ---- Load all 8 B fragments first (one wait, then 8 chained WMMAs)
        v16h Breg[8];
        const _Float16* bp = lds + (size_t)(cur * 16 + l15) * LDSS + half * 8;
#pragma unroll
        for (int ks = 0; ks < 8; ks++) {
            h8 lo = *(const h8*)(bp + ks * 32);
            h8 hi = *(const h8*)(bp + ks * 32 + 16);
            Breg[ks] = __builtin_shufflevector(lo, hi, 0,1,2,3,4,5,6,7,
                                                        8,9,10,11,12,13,14,15);
        }
        v8f acc = {};
#pragma unroll
        for (int ks = 0; ks < 8; ks++) {
            acc = __builtin_amdgcn_wmma_f32_16x16x32_f16(
                      false, Areg[ks], false, Breg[ks], (short)0, acc, false, false);
        }

        // ---- Streaming softmax stats. C layout: VGPR r holds M = m0+r+half*8,
        // N = j0 + l15 for this lane.
        const int col = j0 + l15;
#pragma unroll
        for (int r = 0; r < 8; r++) {
            const int row = m0 + r + half * 8;
            const float s = acc[r] * TEMP_INV;
            const float e = __expf(s - MAXLOGIT);
            sume[r] += (col == row) ? 0.0f : e;   // diagonal masked (exp(-1e9)=0)
            const int tgt = (row < BATCH) ? row + BATCH : row - BATCH;
            if (col == tgt) tg[r] = s;
        }
    }

    // ---- Reduce across the 16 lanes of each half; lanes 0 & 16 commit.
#pragma unroll
    for (int r = 0; r < 8; r++) {
        float se = sume[r], tt = tg[r];
#pragma unroll
        for (int m = 8; m >= 1; m >>= 1) {
            se += __shfl_xor(se, m, 32);
            tt += __shfl_xor(tt, m, 32);
        }
        if (l15 == 0) {
            const int row = m0 + r + half * 8;
            atomicAdd(&S[row], se);
            atomicAdd(&T[row], tt);
        }
    }
}

// ---------------- Kernel 3: final loss reduction ----------------------------
__global__ void __launch_bounds__(256) loss_kernel(const float* __restrict__ S,
                                                   const float* __restrict__ T,
                                                   float* __restrict__ out) {
    const int i = blockIdx.x * 256 + threadIdx.x;
    // loss_i = -(t_i - MAX - log(S_i)) = MAX + log(S_i) - t_i
    float v = MAXLOGIT + logf(S[i]) - T[i];
#pragma unroll
    for (int m = 16; m >= 1; m >>= 1) v += __shfl_xor(v, m, 32);
    if ((threadIdx.x & 31) == 0) atomicAdd(out, v * (1.0f / NROWS));
}

// ---------------------------------------------------------------------------
extern "C" void kernel_launch(void* const* d_in, const int* in_sizes, int n_in,
                              void* d_out, int out_size, void* d_ws, size_t ws_size,
                              hipStream_t stream) {
    const float* zi = (const float*)d_in[0];
    const float* zj = (const float*)d_in[1];
    float* out = (float*)d_out;

    _Float16* zn = (_Float16*)d_ws;                                  // 4 MB
    float* S = (float*)((char*)d_ws + (size_t)NROWS * DIM * 2);      // 32 KB
    float* T = S + NROWS;                                            // 32 KB

    hipMemsetAsync(S, 0, 2 * NROWS * sizeof(float), stream);
    hipMemsetAsync(out, 0, sizeof(float), stream);

    nrm_kernel<<<NROWS / 8, 256, 0, stream>>>(zi, zj, zn);

    dim3 grid(NROWS / (8 * 16), NSPLIT);   // (64, 8) = 512 workgroups
    simk_kernel<<<grid, 256, 0, stream>>>(zn, S, T);

    loss_kernel<<<NROWS / 256, 256, 0, stream>>>(S, T, out);
}
